// AttentionWithOutput_49563922596237
// MI455X (gfx1250) — matrix operand order repaired and checked by reading, not compiled, not run on hardware
//
#include <hip/hip_runtime.h>

typedef __attribute__((ext_vector_type(16))) _Float16 v16h;
typedef __attribute__((ext_vector_type(8)))  float    v8f;

// ---------------------------------------------------------------------------
// WMMA helpers (CDNA5 wave32, D = A(16x32,f16) * B(32x16,f16) + C(16x16,f32))
// ---------------------------------------------------------------------------
__device__ __forceinline__ v8f wmma_f16(v16h a, v16h b, v8f c) {
  return __builtin_amdgcn_wmma_f32_16x16x32_f16(
      /*neg_a=*/false, a, /*neg_b=*/false, b,
      /*c_mod=*/(short)0, c, /*reuse_a=*/false, /*reuse_b=*/false);
}

// A fragment (16x32 f16, row-major source, leading dim `ld`):
// lane: m = lane&15, half = lane>>4 ; VGPR j holds K = ((j&4)<<2) + 2*(j&3) + 8*half
__device__ __forceinline__ v16h load_afrag(const _Float16* base, int ld, int lane) {
  const int m = lane & 15, half = lane >> 4;
  union { v16h v; unsigned int u[8]; } r;
  const _Float16* p = base + m * ld + 8 * half;
#pragma unroll
  for (int j = 0; j < 8; ++j) {
    const int k = ((j & 4) << 2) + 2 * (j & 3);
    r.u[j] = *(const unsigned int*)(p + k);
  }
  return r.v;
}

// B fragment (32x16 f16) from a TRANSPOSED tile Bt[n][k] (leading dim `ld`):
// lane: n = lane&15, half = lane>>4 ; VGPR j holds K = ((j&4)<<2) + 2*(j&3) + 8*half
__device__ __forceinline__ v16h load_bfrag_t(const _Float16* base, int ld, int lane) {
  const int n = lane & 15, half = lane >> 4;
  union { v16h v; unsigned int u[8]; } r;
  const _Float16* p = base + n * ld + 8 * half;
#pragma unroll
  for (int j = 0; j < 8; ++j) r.u[j] = *(const unsigned int*)(p + ((j & 4) << 2) + 2 * (j & 3));
  return r.v;
}

// ---------------------------------------------------------------------------
// GEMM: 128x128 block tile, BK=32, 256 threads = 8 waves, each wave 32x64.
// Double-buffered LDS, one barrier per K-step, packed f16 LDS stores.
// MODE 0: QKV epilogue -> f16 q(scaled)/k/v in [B,H,N,D] layout
// MODE 1: proj epilogue -> fp32 out with bias
// ---------------------------------------------------------------------------
constexpr int BM = 128, BN = 128, BK = 32, BKP = BK + 8;

template <int MODE>
__global__ __launch_bounds__(256) void gemm_f16w(
    const float* __restrict__ A, const float* __restrict__ W,
    const float* __restrict__ bias, int K, int Ncols,
    _Float16* __restrict__ qg, _Float16* __restrict__ kg,
    _Float16* __restrict__ vg, float* __restrict__ outF) {
  __shared__ __align__(16) _Float16 As[2][BM][BKP];
  __shared__ __align__(16) _Float16 Bs[2][BN][BKP];

  const int tid = threadIdx.x;
  const int lane = tid & 31, wid = tid >> 5;
  const int wm = wid & 3, wn = wid >> 2;              // 4x2 wave grid: 32 rows x 64 cols
  const int m0 = blockIdx.y * BM, n0 = blockIdx.x * BN;

  const v8f vzero = {};
  v8f acc[2][4];
#pragma unroll
  for (int a = 0; a < 2; ++a)
#pragma unroll
    for (int b = 0; b < 4; ++b) acc[a][b] = vzero;

  const int arow = tid >> 3, acol = (tid & 7) * 4;    // A loader: 128x32 via float4
  const int bn = tid & 127, bk = (tid >> 7) * 2;      // B loader: 32x128 as k-pairs

  float4 aR[4];
  float  bR[8][2];

  auto gload = [&](int k0) {
#pragma unroll
    for (int i = 0; i < 4; ++i)
      aR[i] = *(const float4*)(A + (size_t)(m0 + arow + 32 * i) * K + k0 + acol);
#pragma unroll
    for (int i = 0; i < 8; ++i) {
      const int kr = bk + 4 * i;
      bR[i][0] = W[(size_t)(k0 + kr + 0) * Ncols + n0 + bn];
      bR[i][1] = W[(size_t)(k0 + kr + 1) * Ncols + n0 + bn];
    }
  };
  auto sstore = [&](int buf) {
#pragma unroll
    for (int i = 0; i < 4; ++i) {           // packed 4xf16 = one b64 store
      union { _Float16 h[4]; unsigned long long u; } p;
      p.h[0] = (_Float16)aR[i].x; p.h[1] = (_Float16)aR[i].y;
      p.h[2] = (_Float16)aR[i].z; p.h[3] = (_Float16)aR[i].w;
      *(unsigned long long*)&As[buf][arow + 32 * i][acol] = p.u;
    }
#pragma unroll
    for (int i = 0; i < 8; ++i) {           // packed 2xf16 = one b32 store (transposed)
      union { _Float16 h[2]; unsigned int u; } p;
      p.h[0] = (_Float16)bR[i][0]; p.h[1] = (_Float16)bR[i][1];
      *(unsigned int*)&Bs[buf][bn][bk + 4 * i] = p.u;
    }
  };

  gload(0);
  sstore(0);
  __syncthreads();

  int buf = 0;
  for (int k0 = 0; k0 < K; k0 += BK) {
    const bool last = (k0 + BK >= K);
    if (!last) gload(k0 + BK);              // prefetch next tile into registers

    const v16h a0 = load_afrag(&As[buf][wm * 32 + 0][0], BKP, lane);
    const v16h a1 = load_afrag(&As[buf][wm * 32 + 16][0], BKP, lane);
#pragma unroll
    for (int bt = 0; bt < 4; ++bt) {
      const v16h bb = load_bfrag_t(&Bs[buf][wn * 64 + bt * 16][0], BKP, lane);
      acc[0][bt] = wmma_f16(a0, bb, acc[0][bt]);
      acc[1][bt] = wmma_f16(a1, bb, acc[1][bt]);
    }

    if (!last) {                            // fill ping-pong buffer, single barrier
      sstore(buf ^ 1);
      __syncthreads();
      buf ^= 1;
    }
  }

  const int half = lane >> 4, nn = lane & 15;
#pragma unroll
  for (int a = 0; a < 2; ++a) {
#pragma unroll
    for (int bt = 0; bt < 4; ++bt) {
      const int col = n0 + wn * 64 + bt * 16 + nn;
      const float bv = bias[col];
#pragma unroll
      for (int r = 0; r < 8; ++r) {
        const int row = m0 + wm * 32 + a * 16 + r + 8 * half;
        const float val = acc[a][bt][r] + bv;
        if constexpr (MODE == 0) {
          const int which = col / 768;
          const int c = col % 768;
          const int h = c >> 6, d = c & 63;
          const int bb2 = row >> 11, n = row & 2047;
          const size_t idx = (((size_t)(bb2 * 12 + h)) * 2048 + n) * 64 + d;
          if (which == 0)      qg[idx] = (_Float16)(val * 0.125f);  // D^-0.5
          else if (which == 1) kg[idx] = (_Float16)val;
          else                 vg[idx] = (_Float16)val;
        } else {
          outF[(size_t)row * Ncols + col] = val;
        }
      }
    }
  }
}

// ---------------------------------------------------------------------------
// Flash attention: block = 128 Q rows of one (b,h); 8 waves x 16 rows each.
// Key tiles of 64; K tile row-major in LDS, V tile transposed in LDS.
// ---------------------------------------------------------------------------
__global__ __launch_bounds__(256) void attn_fwd(
    const _Float16* __restrict__ qg, const _Float16* __restrict__ kg,
    const _Float16* __restrict__ vg, float* __restrict__ attn_out) {
  __shared__ __align__(16) _Float16 Ks[64][72];      // [key][d]
  __shared__ __align__(16) _Float16 Vt[64][72];      // [d][key]
  __shared__ __align__(16) _Float16 Pl[8][16][72];   // per-wave P scratch

  const int tid = threadIdx.x, lane = tid & 31, wid = tid >> 5;
  const int bh = blockIdx.y;                  // b*12 + h
  const int b = bh / 12, h = bh % 12;
  const int row0 = blockIdx.x * 128;
  const size_t base = (size_t)bh * 2048 * 64;
  const int half = lane >> 4, nn = lane & 15;

  // per-wave Q fragments (16 rows x 64), held in registers for whole kernel
  const _Float16* qrow = qg + base + (size_t)(row0 + wid * 16) * 64;
  const v16h qa0 = load_afrag(qrow + 0, 64, lane);
  const v16h qa1 = load_afrag(qrow + 32, 64, lane);

  const v8f vzero = {};
  float mst[8], lst[8];
  v8f o[4];
#pragma unroll
  for (int r = 0; r < 8; ++r) { mst[r] = -3.0e38f; lst[r] = 0.f; }
#pragma unroll
  for (int d = 0; d < 4; ++d) o[d] = vzero;

  const int lkey = tid >> 2;          // 0..63
  const int ldc = (tid & 3) * 16;     // 0,16,32,48

  for (int key0 = 0; key0 < 2048; key0 += 64) {
    __syncthreads();
    // K tile: [64 keys][64 d], 16B vector loads
#pragma unroll
    for (int c = 0; c < 2; ++c) {
      const int d = ldc + 8 * c;
      *(int4*)&Ks[lkey][d] = *(const int4*)(kg + base + (size_t)(key0 + lkey) * 64 + d);
    }
    // V tile transposed: Vt[d][key]
#pragma unroll
    for (int c = 0; c < 8; ++c) {
      const int d = ldc + 2 * c;
      union { unsigned int u; _Float16 hh[2]; } cv;
      cv.u = *(const unsigned int*)(vg + base + (size_t)(key0 + lkey) * 64 + d);
      Vt[d + 0][lkey] = cv.hh[0];
      Vt[d + 1][lkey] = cv.hh[1];
    }
    // prefetch next tile into L2/WGP$ (overlaps with softmax VALU phase)
    if (key0 + 64 < 2048) {
      __builtin_prefetch(kg + base + (size_t)(key0 + 64 + lkey) * 64 + ldc, 0, 1);
      __builtin_prefetch(vg + base + (size_t)(key0 + 64 + lkey) * 64 + ldc, 0, 1);
    }
    __syncthreads();

    // S = Q @ K^T : 16 rows x 64 keys, 8 WMMAs
    v8f s[4];
#pragma unroll
    for (int sub = 0; sub < 4; ++sub) {
      v8f z = vzero;
      const v16h bk0 = load_bfrag_t(&Ks[sub * 16][0], 72, lane);
      const v16h bk1 = load_bfrag_t(&Ks[sub * 16][32], 72, lane);
      z = wmma_f16(qa0, bk0, z);
      z = wmma_f16(qa1, bk1, z);
      s[sub] = z;
    }

    // online softmax (rows r+8*half, duplicated across 16-lane halves)
#pragma unroll
    for (int r = 0; r < 8; ++r) {
      float rowmax = fmaxf(fmaxf(s[0][r], s[1][r]), fmaxf(s[2][r], s[3][r]));
#pragma unroll
      for (int off = 1; off < 16; off <<= 1)
        rowmax = fmaxf(rowmax, __shfl_xor(rowmax, off, 32));
      const float mnew = fmaxf(mst[r], rowmax);
      const float scale = __expf(mst[r] - mnew);
      float lsum = 0.f;
#pragma unroll
      for (int sub = 0; sub < 4; ++sub) {
        const float p = __expf(s[sub][r] - mnew);
        Pl[wid][r + 8 * half][sub * 16 + nn] = (_Float16)p;
        lsum += p;
      }
#pragma unroll
      for (int off = 1; off < 16; off <<= 1)
        lsum += __shfl_xor(lsum, off, 32);
      lst[r] = lst[r] * scale + lsum;
      mst[r] = mnew;
#pragma unroll
      for (int d = 0; d < 4; ++d) o[d][r] *= scale;
    }
    __syncthreads();  // P visible (C-layout -> A-layout via LDS)

    // O += P @ V : 8 WMMAs
    const v16h pa0 = load_afrag(&Pl[wid][0][0], 72, lane);
    const v16h pa1 = load_afrag(&Pl[wid][0][32], 72, lane);
#pragma unroll
    for (int dsub = 0; dsub < 4; ++dsub) {
      const v16h vb0 = load_bfrag_t(&Vt[dsub * 16][0], 72, lane);
      const v16h vb1 = load_bfrag_t(&Vt[dsub * 16][32], 72, lane);
      o[dsub] = wmma_f16(pa0, vb0, o[dsub]);
      o[dsub] = wmma_f16(pa1, vb1, o[dsub]);
    }
  }

  // write ctx -> attn_out in [B, N, H*64] layout
#pragma unroll
  for (int dsub = 0; dsub < 4; ++dsub) {
#pragma unroll
    for (int r = 0; r < 8; ++r) {
      const int nrow = row0 + wid * 16 + r + 8 * half;
      const int cch = h * 64 + dsub * 16 + nn;
      attn_out[((size_t)b * 2048 + nrow) * 768 + cch] = o[dsub][r] / lst[r];
    }
  }
}

// ---------------------------------------------------------------------------
// Launch: QKV GEMM -> flash attention -> proj GEMM
// ---------------------------------------------------------------------------
extern "C" void kernel_launch(void* const* d_in, const int* in_sizes, int n_in,
                              void* d_out, int out_size, void* d_ws, size_t ws_size,
                              hipStream_t stream) {
  (void)in_sizes; (void)n_in; (void)out_size; (void)ws_size;
  const float* x     = (const float*)d_in[0];  // [4,2048,768]
  const float* Wqkv  = (const float*)d_in[1];  // [768,2304]
  const float* bqkv  = (const float*)d_in[2];  // [2304]
  const float* Wproj = (const float*)d_in[3];  // [768,768]
  const float* bproj = (const float*)d_in[4];  // [768]

  const size_t BNC = (size_t)4 * 2048 * 768;   // 6,291,456
  float* out      = (float*)d_out;             // [4,2048,768]
  float* attn_out = out + BNC;                 // [4,2048,768]

  _Float16* qws = (_Float16*)d_ws;             // [B,H,N,D] f16, pre-scaled
  _Float16* kws = qws + BNC;
  _Float16* vws = kws + BNC;

  // 1) qkv = x @ W_qkv + b  -> q/k/v f16 in [B,H,N,D]
  gemm_f16w<0><<<dim3(2304 / BN, 8192 / BM), 256, 0, stream>>>(
      x, Wqkv, bqkv, 768, 2304, qws, kws, vws, nullptr);

  // 2) flash attention -> attn_out [B,N,C]
  attn_fwd<<<dim3(2048 / 128, 48), 256, 0, stream>>>(qws, kws, vws, attn_out);

  // 3) out = attn_out @ W_proj + b
  gemm_f16w<1><<<dim3(768 / BN, 8192 / BM), 256, 0, stream>>>(
      attn_out, Wproj, bproj, 768, 768, nullptr, nullptr, nullptr, out);
}
